// RNN_Model_74998718922794
// MI455X (gfx1250) — matrix operand order repaired and checked by reading, not compiled, not run on hardware
//
#include <hip/hip_runtime.h>
#include <math.h>

// ---------------------------------------------------------------------------
// Types for gfx1250 WMMA
// ---------------------------------------------------------------------------
typedef __attribute__((ext_vector_type(16))) __bf16 v16bf;
typedef __attribute__((ext_vector_type(8)))  float  v8f;

union Frag16 { v16bf v; uint4 q[2]; };

// A-fragment (16x32 bf16, M-rows) from LDS, row-major, stride in elements.
// Contiguous k order used here (not ISA table order): lane holds row (lane&15); lanes 0-15 K={0..15}, lanes 16-31 K={16..31}.
__device__ inline v16bf frag_a_lds(const __bf16* base, int stride, int lane, int kBase) {
    int row = lane & 15;
    int c0  = kBase + ((lane >> 4) << 4);
    const __bf16* p = base + row * stride + c0;
    Frag16 f;
    f.q[0] = *(const uint4*)(p);
    f.q[1] = *(const uint4*)(p + 8);
    return f.v;
}

// B-fragment (32x16 bf16, K x N) built from row-major W[N][K] (i.e. B = W^T).
__device__ inline v16bf frag_b_global(const __bf16* W, int ldk, int n0, int lane, int kBase) {
    int col = lane & 15;
    int k0  = kBase + ((lane >> 4) << 4);
    const __bf16* p = W + (size_t)(n0 + col) * (size_t)ldk + k0;
    Frag16 f;
    f.q[0] = *(const uint4*)(p);
    f.q[1] = *(const uint4*)(p + 8);
    return f.v;
}

__device__ inline v16bf frag_b_lds(const __bf16* base, int stride, int lane) {
    int row = lane & 15;
    int c0  = (lane >> 4) << 4;
    const __bf16* p = base + row * stride + c0;
    Frag16 f;
    f.q[0] = *(const uint4*)(p);
    f.q[1] = *(const uint4*)(p + 8);
    return f.v;
}

__device__ inline v8f wmma_bf16(v16bf a, v16bf b, v8f c) {
    return __builtin_amdgcn_wmma_f32_16x16x32_bf16(false, a, false, b, (short)0, c, false, false);
}

// Async Global -> LDS copy (16B), tracked by ASYNCcnt. LDS address is the low
// 32 bits of the generic pointer (ISA 10.2).
__device__ inline void async_cp_b128(const __bf16* g, const __bf16* s) {
    unsigned lds = (unsigned)(uintptr_t)(const void*)s;
    asm volatile("global_load_async_to_lds_b128 %0, %1, off"
                 :: "v"(lds), "v"(g) : "memory");
}
__device__ inline void wait_async0() {
    asm volatile("s_wait_asynccnt 0x0" ::: "memory");
}

// ---------------------------------------------------------------------------
// fp32 -> bf16 conversion (weights)
// ---------------------------------------------------------------------------
__global__ void k_cvt_bf16(const float* __restrict__ src, __bf16* __restrict__ dst, int n) {
    for (int i = blockIdx.x * blockDim.x + threadIdx.x; i < n; i += gridDim.x * blockDim.x)
        dst[i] = (__bf16)src[i];
}

// ---------------------------------------------------------------------------
// Embedding gather -> bf16
// ---------------------------------------------------------------------------
__global__ void k_embed(const int* __restrict__ tokens, const float* __restrict__ emb,
                        __bf16* __restrict__ x0) {
    int row = blockIdx.x;                 // b*L + l  (8192 rows)
    int tok = tokens[row];
    const float* src = emb + (size_t)tok * 512;
    __bf16* dst = x0 + (size_t)row * 512;
    for (int i = threadIdx.x; i < 512; i += blockDim.x)
        dst[i] = (__bf16)src[i];
}

// ---------------------------------------------------------------------------
// Generic GEMM:  C[M,N] (f32) = A[M,K](bf16) * B[N,K](bf16)^T + bias1 (+bias2)
// Block tile 128x128, BK=32, 8 waves of 2(M)x4(N) 16x16 WMMA tiles.
// Double-buffered LDS filled by async global->LDS copies (ASYNCcnt),
// one barrier per K-step.
// ---------------------------------------------------------------------------
#define GLDA 40   // padded LDS stride (elements) -> conflict-free strided frag reads
#define GBUF (128 * GLDA)

__global__ __launch_bounds__(256) void k_gemm(const __bf16* __restrict__ A,
                                              const __bf16* __restrict__ B,
                                              const float* __restrict__ bias1,
                                              const float* __restrict__ bias2,
                                              float* __restrict__ C,
                                              int M, int N, int K) {
    __shared__ alignas(16) __bf16 sA[2 * GBUF];
    __shared__ alignas(16) __bf16 sB[2 * GBUF];

    const int n0 = blockIdx.x * 128;
    const int m0 = blockIdx.y * 128;
    const int tid  = threadIdx.x;
    const int lane = tid & 31;
    const int w    = tid >> 5;           // 0..7
    const int wm   = w & 3;              // 4 M-strips of 32
    const int wn   = w >> 2;             // 2 N-strips of 64

    v8f zero = {};
    v8f acc[2][4];
#pragma unroll
    for (int i = 0; i < 2; ++i)
#pragma unroll
        for (int j = 0; j < 4; ++j) acc[i][j] = zero;

    // staging: each thread moves 32B of A and 32B of B per K-step (4 async b128)
    const int r  = tid >> 1;             // 0..127
    const int cc = (tid & 1) << 4;       // 0 or 16
    const __bf16* gA = A + (size_t)(m0 + r) * (size_t)K + cc;
    int gn = n0 + r; if (gn >= N) gn = N - 1;
    const __bf16* gB = B + (size_t)gn * (size_t)K + cc;

    // prologue: fill buffer 0
    {
        __bf16* dA = sA + r * GLDA + cc;
        __bf16* dB = sB + r * GLDA + cc;
        async_cp_b128(gA,     dA);
        async_cp_b128(gA + 8, dA + 8);
        async_cp_b128(gB,     dB);
        async_cp_b128(gB + 8, dB + 8);
    }
    wait_async0();
    __syncthreads();

    int buf = 0;
    for (int k0 = 0; k0 < K; k0 += 32, buf ^= 1) {
        // prefetch next K-tile into the other buffer while we compute
        if (k0 + 32 < K) {
            __bf16* dA = sA + (buf ^ 1) * GBUF + r * GLDA + cc;
            __bf16* dB = sB + (buf ^ 1) * GBUF + r * GLDA + cc;
            async_cp_b128(gA + k0 + 32,     dA);
            async_cp_b128(gA + k0 + 32 + 8, dA + 8);
            async_cp_b128(gB + k0 + 32,     dB);
            async_cp_b128(gB + k0 + 32 + 8, dB + 8);
        }

        const __bf16* pA = sA + buf * GBUF;
        const __bf16* pB = sB + buf * GBUF;
        v16bf a0 = frag_a_lds(pA + (wm * 32 +  0) * GLDA, GLDA, lane, 0);
        v16bf a1 = frag_a_lds(pA + (wm * 32 + 16) * GLDA, GLDA, lane, 0);
#pragma unroll
        for (int tn = 0; tn < 4; ++tn) {
            v16bf b = frag_b_lds(pB + (wn * 64 + tn * 16) * GLDA, GLDA, lane);
            acc[0][tn] = wmma_bf16(a0, b, acc[0][tn]);
            acc[1][tn] = wmma_bf16(a1, b, acc[1][tn]);
        }

        wait_async0();       // our prefetch into buf^1 landed in LDS
        __syncthreads();     // everyone done reading buf; safe to overwrite next iter
    }

    // Epilogue: D layout — VGPR r: M = r + 8*(lane>=16), N = lane&15
#pragma unroll
    for (int tm = 0; tm < 2; ++tm)
#pragma unroll
        for (int tn = 0; tn < 4; ++tn)
#pragma unroll
            for (int rr = 0; rr < 8; ++rr) {
                int m = m0 + wm * 32 + tm * 16 + rr + ((lane >> 4) << 3);
                int n = n0 + wn * 64 + tn * 16 + (lane & 15);
                if (n < N) {
                    float bb = bias1[n] + (bias2 ? bias2[n] : 0.0f);
                    C[(size_t)m * (size_t)N + n] = acc[tm][tn][rr] + bb;
                }
            }
}

// ---------------------------------------------------------------------------
// RNN recurrence, one direction half-batch per block (batch rows independent!)
// grid = 4:  blockIdx -> dir = bx>>1, mhalf = bx&1 (16 batch rows each).
// h (16x768 bf16) lives in LDS; per step: 48 N-tiles x 24 K-chunks of WMMA.
// ---------------------------------------------------------------------------
#define HSTR 776  // 768 + 8 pad -> conflict-free A-frag reads

__global__ __launch_bounds__(512) void k_rnn(const float* __restrict__ xp,
                                             const __bf16* __restrict__ whh,
                                             __bf16* __restrict__ out) {
    const int dir = blockIdx.x >> 1;
    const int mh  = blockIdx.x & 1;
    const __bf16* W = whh + (size_t)dir * 768 * 768;

    __shared__ alignas(16) __bf16 h[16 * HSTR];
    for (int i = threadIdx.x; i < 16 * HSTR; i += blockDim.x) h[i] = (__bf16)0.0f;
    __syncthreads();

    const int lane = threadIdx.x & 31;
    const int w    = threadIdx.x >> 5;   // 0..15 waves, 3 N-tiles each
    v8f zero = {};

    for (int t = 0; t < 256; ++t) {
        const int l = dir ? (255 - t) : t;

        v8f acc[3];
        acc[0] = zero; acc[1] = zero; acc[2] = zero;
        for (int kc = 0; kc < 24; ++kc) {
            v16bf aF = frag_a_lds(h, HSTR, lane, kc * 32);
#pragma unroll
            for (int i = 0; i < 3; ++i) {
                int tn = w * 3 + i;
                v16bf bF = frag_b_global(W, 768, tn * 16, lane, kc * 32);
                acc[i] = wmma_bf16(aF, bF, acc[i]);
            }
        }

        __bf16 hnew[3][8];
#pragma unroll
        for (int i = 0; i < 3; ++i) {
            int tn = w * 3 + i;
#pragma unroll
            for (int rr = 0; rr < 8; ++rr) {
                int m = rr + ((lane >> 4) << 3);          // row within block (0..15)
                int b = mh * 16 + m;
                int n = tn * 16 + (lane & 15);
                size_t idx = ((size_t)b * 256 + l) * 1536 + dir * 768 + n;
                float v = tanhf(xp[idx] + acc[i][rr]);
                __bf16 hb = (__bf16)v;
                out[idx] = hb;
                hnew[i][rr] = hb;
            }
        }
        __syncthreads();   // all reads of old h done
#pragma unroll
        for (int i = 0; i < 3; ++i) {
            int tn = w * 3 + i;
#pragma unroll
            for (int rr = 0; rr < 8; ++rr) {
                int m = rr + ((lane >> 4) << 3);
                h[m * HSTR + tn * 16 + (lane & 15)] = hnew[i][rr];
            }
        }
        __syncthreads();   // h fully updated
    }
}

// ---------------------------------------------------------------------------
// BatchNorm over axes (B, channels) per timestep l. One block per l.
// ---------------------------------------------------------------------------
__global__ __launch_bounds__(256) void k_bn_stats(const __bf16* __restrict__ x,
                                                  const float* __restrict__ gamma,
                                                  const float* __restrict__ beta,
                                                  float* __restrict__ ss) {
    const int l = blockIdx.x;
    float s = 0.0f, sq = 0.0f;
    for (int j = threadIdx.x; j < 32 * 1536; j += blockDim.x) {
        int b = j / 1536, c = j - b * 1536;
        float v = (float)x[((size_t)b * 256 + l) * 1536 + c];
        s += v; sq += v * v;
    }
    __shared__ float rs[256], rq[256];
    rs[threadIdx.x] = s; rq[threadIdx.x] = sq;
    __syncthreads();
    for (int k = 128; k > 0; k >>= 1) {
        if (threadIdx.x < k) {
            rs[threadIdx.x] += rs[threadIdx.x + k];
            rq[threadIdx.x] += rq[threadIdx.x + k];
        }
        __syncthreads();
    }
    if (threadIdx.x == 0) {
        const float inv = 1.0f / (32.0f * 1536.0f);
        float mean = rs[0] * inv;
        float var  = rq[0] * inv - mean * mean;
        float scale = gamma[l] * rsqrtf(var + 1e-5f);
        ss[2 * l + 0] = scale;
        ss[2 * l + 1] = beta[l] - mean * scale;
    }
}

__global__ void k_bn_apply(__bf16* __restrict__ x, const float* __restrict__ ss) {
    const int total = 8192 * 1536;
    for (int i = blockIdx.x * blockDim.x + threadIdx.x; i < total; i += gridDim.x * blockDim.x) {
        int l = (i / 1536) & 255;
        float v = (float)x[i] * ss[2 * l] + ss[2 * l + 1];
        x[i] = (__bf16)v;
    }
}

// ---------------------------------------------------------------------------
// Host launcher
// ---------------------------------------------------------------------------
extern "C" void kernel_launch(void* const* d_in, const int* in_sizes, int n_in,
                              void* d_out, int out_size, void* d_ws, size_t ws_size,
                              hipStream_t stream) {
    (void)in_sizes; (void)n_in; (void)out_size; (void)ws_size;

    const int*   tokens  = (const int*)  d_in[0];
    const float* emb     = (const float*)d_in[1];
    const float* w_ih_l0 = (const float*)d_in[2];
    const float* w_hh_l0 = (const float*)d_in[3];
    const float* b_ih_l0 = (const float*)d_in[4];
    const float* b_hh_l0 = (const float*)d_in[5];
    const float* w_ih_l1 = (const float*)d_in[6];
    const float* w_hh_l1 = (const float*)d_in[7];
    const float* b_ih_l1 = (const float*)d_in[8];
    const float* b_hh_l1 = (const float*)d_in[9];
    const float* gamma   = (const float*)d_in[10];
    const float* beta    = (const float*)d_in[11];
    const float* lin_w   = (const float*)d_in[12];
    const float* lin_b   = (const float*)d_in[13];
    float* out = (float*)d_out;

    // workspace carve-up
    char* base = (char*)d_ws;
    size_t off = 0;
    auto carve = [&](size_t bytes) -> char* {
        char* p = base + off;
        off += (bytes + 255) & ~(size_t)255;
        return p;
    };
    __bf16* x0    = (__bf16*)carve((size_t)8192 * 512 * 2);
    __bf16* wih0b = (__bf16*)carve((size_t)1536 * 512 * 2);
    __bf16* whh0b = (__bf16*)carve((size_t)1536 * 768 * 2);
    __bf16* wih1b = (__bf16*)carve((size_t)1536 * 1536 * 2);
    __bf16* whh1b = (__bf16*)carve((size_t)1536 * 768 * 2);
    __bf16* linwb = (__bf16*)carve((size_t)10000 * 1536 * 2);
    float*  xp    = (float*) carve((size_t)8192 * 1536 * 4);
    __bf16* out0  = (__bf16*)carve((size_t)8192 * 1536 * 2);
    __bf16* out1  = (__bf16*)carve((size_t)8192 * 1536 * 2);
    float*  ss    = (float*) carve((size_t)256 * 2 * 4);

    // 1) weight conversions to bf16
    k_cvt_bf16<<<2048, 256, 0, stream>>>(w_ih_l0, wih0b, 1536 * 512);
    k_cvt_bf16<<<2048, 256, 0, stream>>>(w_hh_l0, whh0b, 1536 * 768);
    k_cvt_bf16<<<2048, 256, 0, stream>>>(w_ih_l1, wih1b, 1536 * 1536);
    k_cvt_bf16<<<2048, 256, 0, stream>>>(w_hh_l1, whh1b, 1536 * 768);
    k_cvt_bf16<<<4096, 256, 0, stream>>>(lin_w,   linwb, 10000 * 1536);

    // 2) embedding gather
    k_embed<<<8192, 128, 0, stream>>>(tokens, emb, x0);

    // 3) layer 0 input projection: xp = x0 @ [wih0_d0; wih0_d1]^T + b_ih + b_hh
    k_gemm<<<dim3(12, 64), 256, 0, stream>>>(x0, wih0b, b_ih_l0, b_hh_l0, xp, 8192, 1536, 512);

    // 4) layer 0 recurrence (both dirs, batch split into 2 halves => 4 blocks)
    k_rnn<<<4, 512, 0, stream>>>(xp, whh0b, out0);

    // 5) layer 1 input projection
    k_gemm<<<dim3(12, 64), 256, 0, stream>>>(out0, wih1b, b_ih_l1, b_hh_l1, xp, 8192, 1536, 1536);

    // 6) layer 1 recurrence
    k_rnn<<<4, 512, 0, stream>>>(xp, whh1b, out1);

    // 7) batchnorm per timestep, applied in place on out1 (bf16)
    k_bn_stats<<<256, 256, 0, stream>>>(out1, gamma, beta, ss);
    k_bn_apply<<<4096, 256, 0, stream>>>(out1, ss);

    // 8) final vocab GEMM: out = out1 @ lin_w^T + lin_b   (8192 x 10000, K=1536)
    k_gemm<<<dim3(79, 64), 256, 0, stream>>>(out1, linwb, lin_b, nullptr, out, 8192, 10000, 1536);
}
